// GC_WeightNet_11398843203653
// MI455X (gfx1250) — matrix operand order repaired and checked by reading, not compiled, not run on hardware
//
#include <hip/hip_runtime.h>
#include <hip/hip_bf16.h>
#include <math.h>

typedef __attribute__((ext_vector_type(16))) __bf16 v16bf;
typedef __attribute__((ext_vector_type(8)))  float  v8f;

#define C_IN   256
#define C_OUT  256
#define HW     3136
#define WD     56
#define KKDIM  2304      // C_IN * 9, reordered as k = (ib*9 + tap)*32 + il
#define BM     128
#define BN     128
#define BK     32
#define NTILE  25        // ceil(3136/128)
#define LDA    40        // LDS row pitch in bf16 (80B, 16B aligned, bank-rotating)
#define SLABW  244       // 128 + 2*57 halo, padded to multiple of 4
#define SLABN  (32 * SLABW)

// ---------------------------------------------------------------------------
// Kernel A: GC attention pooling + channel-transform MLP -> x_att[b, 256]
// One block per batch sample, 256 threads.
// ---------------------------------------------------------------------------
__global__ __launch_bounds__(256) void gc_attention(
    const float* __restrict__ x, const float* __restrict__ w_mask,
    const float* __restrict__ b_mask, const float* __restrict__ w1,
    const float* __restrict__ b1, const float* __restrict__ ln_g,
    const float* __restrict__ ln_b, const float* __restrict__ w2,
    const float* __restrict__ b2, float* __restrict__ x_att)
{
  __shared__ float lg[HW];
  __shared__ float red[256];
  __shared__ float ctxs[256];
  __shared__ float tl[16];
  __shared__ float stats[2];

  const int tid = threadIdx.x;
  const int b   = blockIdx.x;
  const float* xb = x + (size_t)b * C_IN * HW;

  // logits[hw] = sum_c x[b,c,hw] * w_mask[c]
  float acc[13];
  #pragma unroll
  for (int j = 0; j < 13; ++j) acc[j] = 0.f;
  for (int c = 0; c < C_IN; ++c) {
    const float wv = w_mask[c];
    const float* xr = xb + (size_t)c * HW;
    #pragma unroll
    for (int j = 0; j < 13; ++j) {
      const int p = tid + j * 256;
      if (p < HW) acc[j] = fmaf(xr[p], wv, acc[j]);
    }
  }
  const float bmv = b_mask[0];
  float lmax = -1e30f;
  #pragma unroll
  for (int j = 0; j < 13; ++j) {
    const int p = tid + j * 256;
    if (p < HW) { const float v = acc[j] + bmv; lg[p] = v; lmax = fmaxf(lmax, v); }
  }
  red[tid] = lmax;
  __syncthreads();
  for (int s = 128; s > 0; s >>= 1) {
    if (tid < s) red[tid] = fmaxf(red[tid], red[tid + s]);
    __syncthreads();
  }
  lmax = red[0];
  __syncthreads();

  float lsum = 0.f;
  #pragma unroll
  for (int j = 0; j < 13; ++j) {
    const int p = tid + j * 256;
    if (p < HW) { const float e = __expf(lg[p] - lmax); lg[p] = e; lsum += e; }
  }
  red[tid] = lsum;
  __syncthreads();
  for (int s = 128; s > 0; s >>= 1) {
    if (tid < s) red[tid] += red[tid + s];
    __syncthreads();
  }
  const float inv = 1.f / red[0];
  __syncthreads();

  // ctx[c] = sum_hw x[b,c,hw] * attn[hw]  (thread == channel)
  {
    const float* xr = xb + (size_t)tid * HW;
    float s = 0.f;
    for (int p = 0; p < HW; ++p) s = fmaf(xr[p], lg[p], s);
    ctxs[tid] = s * inv;
  }
  __syncthreads();

  // t = ctx @ w1.T + b1   (R = 16)
  if (tid < 16) {
    float s = b1[tid];
    const float* wr = w1 + tid * C_IN;
    for (int c = 0; c < C_IN; ++c) s = fmaf(ctxs[c], wr[c], s);
    tl[tid] = s;
  }
  __syncthreads();
  if (tid == 0) {
    float mu = 0.f;
    for (int r = 0; r < 16; ++r) mu += tl[r];
    mu *= (1.f / 16.f);
    float var = 0.f;
    for (int r = 0; r < 16; ++r) { const float d = tl[r] - mu; var += d * d; }
    var *= (1.f / 16.f);
    stats[0] = mu;
    stats[1] = rsqrtf(var + 1e-5f);
  }
  __syncthreads();
  if (tid < 16) {
    const float v = (tl[tid] - stats[0]) * stats[1] * ln_g[tid] + ln_b[tid];
    tl[tid] = fmaxf(v, 0.f);
  }
  __syncthreads();
  // x_att = ctx + relu(...) @ w2.T + b2
  {
    float s = ctxs[tid] + b2[tid];
    const float* wr = w2 + tid * 16;
    #pragma unroll
    for (int r = 0; r < 16; ++r) s = fmaf(tl[r], wr[r], s);
    x_att[b * C_OUT + tid] = s;
  }
}

// ---------------------------------------------------------------------------
// Kernel B: reorder w_fc/b_fc from (o, i*9+tap) to k = (ib*9 + tap)*32 + il
// where i = ib*32 + il.  A BK=32 chunk = one tap x one 32-channel block, and
// all 9 taps of a channel block are consecutive (enables LDS slab reuse).
// ---------------------------------------------------------------------------
__global__ __launch_bounds__(256) void reorder_w(
    const float* __restrict__ w_fc, const float* __restrict__ b_fc,
    float* __restrict__ w_r, float* __restrict__ b_r)
{
  const int idx = blockIdx.x * 256 + threadIdx.x;
  if (idx >= C_OUT * KKDIM) return;
  const int o   = idx / KKDIM;
  const int rem = idx % KKDIM;
  const int kt  = rem >> 5;        // 0..71
  const int il  = rem & 31;
  const int ib  = kt / 9;          // channel block 0..7
  const int tap = kt - ib * 9;     // spatial tap 0..8
  const int i   = ib * 32 + il;    // input channel
  const int src = o * KKDIM + i * 9 + tap;
  w_r[idx] = w_fc[src];
  b_r[idx] = b_fc[src];
}

// ---------------------------------------------------------------------------
// Kernel C: per-batch implicit-GEMM conv via bf16 WMMA, f32 accumulate.
// out[b,o,n] = sum_k (x_att[b,o]*W[o,k] + Bf[o,k]) * im2col(x)[b,k,n]
// Per channel block: async-copy a 32ch x 242col f32 slab of x into LDS
// (GLOBAL_LOAD_ASYNC_TO_LDS_B32, ASYNCcnt), then derive all 9 taps' bf16
// B-tiles from the slab. 8 waves/block, each 64x32 = 4x2 WMMA frags.
// ---------------------------------------------------------------------------
__global__ __launch_bounds__(256) void dynconv_wmma(
    const float* __restrict__ x, const float* __restrict__ w_r,
    const float* __restrict__ b_r, const float* __restrict__ x_att,
    float* __restrict__ out)
{
  __shared__ __attribute__((aligned(16))) float  slab[SLABN];   // 32ch x 244
  __shared__ __attribute__((aligned(16))) __bf16 As[BM * LDA];  // A tile, M-major
  __shared__ __attribute__((aligned(16))) __bf16 Xs[BN * LDA];  // X tile, N-major
  __shared__ float satt[BM];

  const int tid   = threadIdx.x;
  const int wave  = tid >> 5;
  const int lane  = tid & 31;
  const int half  = lane >> 4;   // 0: lanes 0-15, 1: lanes 16-31
  const int l16   = lane & 15;
  const int batch = blockIdx.y;
  const int bm    = blockIdx.x / NTILE;
  const int bn    = blockIdx.x % NTILE;
  const int m0    = bm * BM;
  const int n0    = bn * BN;
  const int wm    = (wave & 1) * 64;   // wave tile origin in M
  const int wn    = (wave >> 1) * 32;  // wave tile origin in N

  if (tid < BM) satt[tid] = x_att[batch * C_OUT + m0 + tid];

  // Per-thread staging roles (fixed for whole kernel).
  const int kr = tid >> 3;             // k row / channel-in-block 0..31
  const int c0 = (tid & 7) * 16;       // 16 consecutive n columns
  const int w0 = (n0 + c0) % WD;       // image column of first owned n

  const unsigned long long xbase =
      (unsigned long long)(const void*)(x + (size_t)batch * C_IN * HW);

  v8f acc[4][2];
  #pragma unroll
  for (int fm = 0; fm < 4; ++fm)
    #pragma unroll
    for (int fn = 0; fn < 2; ++fn)
      acc[fm][fn] = (v8f){0.f, 0.f, 0.f, 0.f, 0.f, 0.f, 0.f, 0.f};

  for (int ib = 0; ib < 8; ++ib) {     // channel blocks of 32
    const int i0 = ib * 32;

    // ---- async-stage the f32 x slab: rows i0..i0+31, cols [n0-57, n0+187) ---
    for (int it = 0; it < 31; ++it) {
      const int idx = tid + it * 256;
      if (idx < SLABN) {
        const int ch   = idx / SLABW;
        const int col  = idx - ch * SLABW;
        const int gidx = n0 - 57 + col;
        float* lp = &slab[ch * SLABW + col];
        if (gidx >= 0 && gidx < HW) {
          const unsigned ldsaddr = (unsigned)(size_t)lp;
          const unsigned goff = (unsigned)(((i0 + ch) * HW + gidx) * 4);
          asm volatile("global_load_async_to_lds_b32 %0, %1, %2"
                       :: "v"(ldsaddr), "v"(goff), "s"(xbase) : "memory");
        } else {
          *lp = 0.f;   // zero-fill out-of-tensor halo
        }
      }
    }
    asm volatile("s_wait_asynccnt 0" ::: "memory");
    __syncthreads();

    for (int tap = 0; tap < 9; ++tap) {
      const int kt    = ib * 9 + tap;
      const int dh    = tap / 3 - 1;
      const int dw    = tap % 3 - 1;
      const int shift = 57 + dh * WD + dw;   // slab column shift, 0..114

      __syncthreads();   // previous tap's frag reads done before overwrite

      // Stage A tile: fused x_att[o]*w + b, converted to bf16.
      {
        const int row = tid >> 1;
        const int j0  = (tid & 1) * 16;
        const int o   = m0 + row;
        const float a = satt[row];
        const float* wp = w_r + (size_t)o * KKDIM + kt * BK + j0;
        const float* bp = b_r + (size_t)o * KKDIM + kt * BK + j0;
        __builtin_prefetch(wp + BK, 0, 0);   // next K-chunk of this row
        __bf16* dst = &As[row * LDA + j0];
        #pragma unroll
        for (int j = 0; j < 16; ++j)
          dst[j] = (__bf16)fmaf(a, wp[j], bp[j]);
      }

      // Stage X tile from the LDS slab (shifted window + row-wrap mask).
      {
        const float* srow = &slab[kr * SLABW];
        #pragma unroll
        for (int c = 0; c < 16; ++c) {
          int wc = w0 + c;
          if (wc >= WD) wc -= WD;
          const int wd = wc + dw;
          float v = 0.f;
          if (wd >= 0 && wd < WD) v = srow[c0 + c + shift];
          Xs[(c0 + c) * LDA + kr] = (__bf16)v;
        }
      }

      __syncthreads();

      // Load frags per the 16-bit WMMA VGPR layouts.
      union Frag { v16bf v; uint4 q[2]; };
      Frag afr[4], bfr[2];
      #pragma unroll
      for (int fm = 0; fm < 4; ++fm) {
        const int m = wm + fm * 16 + l16;
        const uint4* p = (const uint4*)&As[m * LDA];
        // lane<16: K {0..7},{16..23}; lane>=16: K {8..15},{24..31}
        afr[fm].q[0] = p[half];
        afr[fm].q[1] = p[2 + half];
      }
      #pragma unroll
      for (int fn = 0; fn < 2; ++fn) {
        const int n = wn + fn * 16 + l16;
        const uint4* p = (const uint4*)&Xs[n * LDA];
        // lane<16: K {0..7},{16..23}; lane>=16: K {8..15},{24..31}
        bfr[fn].q[0] = p[half];
        bfr[fn].q[1] = p[2 + half];
      }

      #pragma unroll
      for (int fm = 0; fm < 4; ++fm)
        #pragma unroll
        for (int fn = 0; fn < 2; ++fn)
          acc[fm][fn] = __builtin_amdgcn_wmma_f32_16x16x32_bf16(
              false, afr[fm].v, false, bfr[fn].v,
              (short)0, acc[fm][fn], false, false);
    }
  }

  // Store: D layout — VGPR v holds row (half*8 + v), lane l16 = column.
  const size_t obase = (size_t)batch * C_OUT * HW;
  #pragma unroll
  for (int fm = 0; fm < 4; ++fm) {
    #pragma unroll
    for (int fn = 0; fn < 2; ++fn) {
      const int n = n0 + wn + fn * 16 + l16;
      if (n < HW) {
        const int orow = m0 + wm + fm * 16 + half * 8;
        float* op = out + obase + (size_t)orow * HW + n;
        #pragma unroll
        for (int v = 0; v < 8; ++v)
          op[(size_t)v * HW] = acc[fm][fn][v];
      }
    }
  }
}

// ---------------------------------------------------------------------------
extern "C" void kernel_launch(void* const* d_in, const int* in_sizes, int n_in,
                              void* d_out, int out_size, void* d_ws, size_t ws_size,
                              hipStream_t stream) {
  const float* x      = (const float*)d_in[0];
  const float* w_mask = (const float*)d_in[1];
  const float* b_mask = (const float*)d_in[2];
  const float* w1     = (const float*)d_in[3];
  const float* b1     = (const float*)d_in[4];
  const float* ln_g   = (const float*)d_in[5];
  const float* ln_b   = (const float*)d_in[6];
  const float* w2     = (const float*)d_in[7];
  const float* b2     = (const float*)d_in[8];
  const float* w_fc   = (const float*)d_in[9];
  const float* b_fc   = (const float*)d_in[10];
  float* out = (float*)d_out;

  // Workspace layout (floats): x_att[32*256] | w_r[256*2304] | b_r[256*2304]
  float* ws    = (float*)d_ws;
  float* x_att = ws;
  float* w_r   = ws + 32 * C_OUT;
  float* b_r   = w_r + (size_t)C_OUT * KKDIM;

  gc_attention<<<32, 256, 0, stream>>>(x, w_mask, b_mask, w1, b1, ln_g, ln_b,
                                       w2, b2, x_att);

  const int nelem = C_OUT * KKDIM;
  reorder_w<<<(nelem + 255) / 256, 256, 0, stream>>>(w_fc, b_fc, w_r, b_r);

  dynconv_wmma<<<dim3(2 * NTILE, 32), 256, 0, stream>>>(x, w_r, b_r, x_att, out);
}
